// GNNWithTransformer_81664508166588
// MI455X (gfx1250) — matrix operand order repaired and checked by reading, not compiled, not run on hardware
//
#include <hip/hip_runtime.h>
#include <math.h>

// ---------------------------------------------------------------------------
// GNN + TransformerConv forward, CDNA5 (gfx1250), wave32, WMMA f16->f32.
//
// v3: software-pipelined GEMM K-loop (double-buffered fragments -> partial
// loadcnt waits, VMEM overlapped with WMMA) and single-pointer B addressing
// with immediate tile offsets (fewer address VALU ops / WAR-hazard NOPs).
// Operands staged to f16: A row-major [M,KP], B swizzled [KP/32][N][32];
// every fragment load is an unconditional aligned global_load_b128.
// Layer-3 NNConv mlp2+einsum fused as one [25600 x 8256] GEMM with a
// register-generated A matrix z[e,k*din+i] = h[e,k]*x[src[e],i].
// ---------------------------------------------------------------------------

typedef __attribute__((ext_vector_type(16))) _Float16 v16h;
typedef __attribute__((ext_vector_type(8)))  _Float16 v8h;
typedef __attribute__((ext_vector_type(8)))  float    v8f;

#define N_GRAPHS 16
#define NPG      40
#define NN       640
#define EE       25600
#define EDGE_DIM 8
#define HDIM     256

__device__ __forceinline__ float gelu_tanh(float x) {
  float x3 = x * x * x;
  return 0.5f * x * (1.0f + tanhf(0.79788456080286535588f * (x + 0.044715f * x3)));
}
__device__ __forceinline__ float wave_sum32(float v) {
#pragma unroll
  for (int off = 16; off > 0; off >>= 1) v += __shfl_xor(v, off, 32);
  return v;
}
__device__ __forceinline__ float wave_max32(float v) {
#pragma unroll
  for (int off = 16; off > 0; off >>= 1) v = fmaxf(v, __shfl_xor(v, off, 32));
  return v;
}
__device__ __forceinline__ int imin(int a, int b) { return a < b ? a : b; }

__device__ __forceinline__ v16h cat8(v8h lo, v8h hi) {
  return __builtin_shufflevector(lo, hi, 0, 1, 2, 3, 4, 5, 6, 7, 8, 9, 10, 11,
                                 12, 13, 14, 15);
}

// ---------------------------------------------------------------------------
// Operand staging.
// pack_a: f32 [M,K] row-major -> f16 [M,KP] row-major, zero-padded K.
// pack_b: f32 [K,N] row-major -> f16 blocks Bp[k0/32][n][32]: element (kk,n)
//         at ((kk>>5)*N + n)*32 + (kk&31).  A lane's 16 B-fragment halves
//         (consecutive K, fixed n) are then 2 contiguous b128 loads.
// ---------------------------------------------------------------------------
__global__ void pack_a(const float* __restrict__ A, _Float16* __restrict__ Ap,
                       int M, int K, int KP) {
  int idx = blockIdx.x * blockDim.x + threadIdx.x;
  if (idx >= M * KP) return;
  int m = idx / KP, kk = idx - m * KP;
  Ap[idx] = (kk < K) ? (_Float16)A[(size_t)m * K + kk] : (_Float16)0.0f;
}
__global__ void pack_b(const float* __restrict__ B, _Float16* __restrict__ Bp,
                       int K, int N, int KP) {
  int idx = blockIdx.x * blockDim.x + threadIdx.x;
  if (idx >= KP * N) return;
  int t = idx & 31;
  int r = idx >> 5;
  int blk = r / N, n = r - blk * N;
  int kk = blk * 32 + t;
  Bp[idx] = (kk < K) ? (_Float16)B[(size_t)kk * N + n] : (_Float16)0.0f;
}

// ---------------------------------------------------------------------------
// GEMM: C[M,N] = act(A16[M,KP] * B16swz + bias).  One wave per 16x(16*NT) C
// tile.  Software-pipelined: fragments for step s+1 are prefetched into a
// separate register set before step s's WMMAs issue, so waits are partial and
// VMEM overlaps matrix math.  Single A & B pointers; B tile offsets are
// compile-time immediates (t*1024 bytes).
// Requires M%16==0, N%(16*NT)==0, KP%32==0.  No guards, no divergence.
// ---------------------------------------------------------------------------
template <int NT>
__global__ void wmma_gemm_f16(const _Float16* __restrict__ A,
                              const _Float16* __restrict__ B,
                              const float* __restrict__ bias,
                              float* __restrict__ C,
                              int M, int N, int KP, int act) {
  int lane = threadIdx.x;
  int row0 = blockIdx.y * 16;
  int col0 = blockIdx.x * (16 * NT);
  int ah = (lane & 16) ? 8 : 0;
  int bk0 = (lane & 16) ? 8 : 0;
  const _Float16* ap = A + (size_t)(row0 + (lane & 15)) * KP + ah;
  const _Float16* bp = B + (size_t)(col0 + (lane & 15)) * 32 + bk0;
  const size_t bstep = (size_t)N * 32;  // halfs per K-step

  v8f zero = {};
  v8f acc[NT];
#pragma unroll
  for (int t = 0; t < NT; ++t) acc[t] = zero;

  v8h a0 = *(const v8h*)ap;
  v8h a1 = *(const v8h*)(ap + 16);
  v8h b0[NT], b1[NT];
#pragma unroll
  for (int t = 0; t < NT; ++t) {
    b0[t] = *(const v8h*)(bp + t * 512);
    b1[t] = *(const v8h*)(bp + t * 512 + 16);
  }

  int nsteps = KP >> 5;
  for (int s = 0; s < nsteps - 1; ++s) {
    v16h a = cat8(a0, a1);
    v16h bt[NT];
#pragma unroll
    for (int t = 0; t < NT; ++t) bt[t] = cat8(b0[t], b1[t]);
    ap += 32;
    bp += bstep;
    a0 = *(const v8h*)ap;
    a1 = *(const v8h*)(ap + 16);
#pragma unroll
    for (int t = 0; t < NT; ++t) {
      b0[t] = *(const v8h*)(bp + t * 512);
      b1[t] = *(const v8h*)(bp + t * 512 + 16);
    }
#pragma unroll
    for (int t = 0; t < NT; ++t)
      acc[t] = __builtin_amdgcn_wmma_f32_16x16x32_f16(false, a, false, bt[t],
                                                      (short)0, acc[t], false,
                                                      false);
  }
  {  // final step (no prefetch)
    v16h a = cat8(a0, a1);
#pragma unroll
    for (int t = 0; t < NT; ++t) {
      v16h b = cat8(b0[t], b1[t]);
      acc[t] = __builtin_amdgcn_wmma_f32_16x16x32_f16(false, a, false, b,
                                                      (short)0, acc[t], false,
                                                      false);
    }
  }

#pragma unroll
  for (int t = 0; t < NT; ++t) {
    int bn = col0 + t * 16 + (lane & 15);
    float bv = bias ? bias[bn] : 0.0f;
#pragma unroll
    for (int r = 0; r < 8; ++r) {
      float val = acc[t][r] + bv;
      if (act == 1) val = gelu_tanh(val);
      C[(size_t)(row0 + r + ah) * N + bn] = val;
    }
  }
}

// ---------------------------------------------------------------------------
// mlp2 weight rearrange -> f16 swizzled blocks [KP/32][DOUT][32]:
// row kk = k*DIN+i -> W2[k][i*DOUT+o] (k < DOUT); DIN bias rows b2[i*DOUT+o];
// zero padding to KP.
// ---------------------------------------------------------------------------
template <int DIN, int DOUT, int KP>
__global__ void prep_w2(const float* __restrict__ W2, const float* __restrict__ b2,
                        _Float16* __restrict__ out) {
  int idx = blockIdx.x * blockDim.x + threadIdx.x;
  if (idx >= KP * DOUT) return;
  int t = idx & 31;
  int r = idx >> 5;
  int blk = r / DOUT, n = r - blk * DOUT;
  int kk = blk * 32 + t;
  float v = 0.0f;
  if (kk < DIN * DOUT) {
    int k = kk / DIN;
    int i = kk - k * DIN;
    v = W2[(size_t)k * (DIN * DOUT) + i * DOUT + n];
  } else if (kk < DIN * DOUT + DIN) {
    v = b2[(kk - DIN * DOUT) * DOUT + n];
  }
  out[idx] = (_Float16)v;
}

// ---------------------------------------------------------------------------
// Fused NNConv message GEMM: msg[E,DOUT] = z @ W2r,
//   z[e, k*DIN+i] = h[e,k]*x[src[e],i]  (k<DOUT), then DIN bias rows = x[..,i].
// B loads are issued FIRST each K-step (single pointer, immediate tile
// offsets) so their latency hides under the VALU-heavy virtual-A generation.
// For DIN>=32 a 32-wide K-step maps to ONE h column (k0 is 32-aligned):
// hv hoisted, x reads contiguous, region select is a uniform branch.
// For DIN=7 indices are clamped (loads always in-bounds) and padded elements
// zeroed via arithmetic select -- no EXEC divergence anywhere.
// ---------------------------------------------------------------------------
template <int DIN, int DOUT, int KP>
__global__ void nnconv_msg(const float* __restrict__ h,      // [E, DOUT]
                           const float* __restrict__ x,      // [N, DIN]
                           const int* __restrict__ src,      // [E]
                           const _Float16* __restrict__ w2s, // [KP/32][DOUT][32]
                           float* __restrict__ msg) {        // [E, DOUT]
  constexpr int NT = DOUT / 16;
  int lane = threadIdx.x;
  int e0 = blockIdx.x * 16;
  int em = e0 + (lane & 15);
  int ah = (lane & 16) ? 8 : 0;
  int bk0 = (lane & 16) ? 8 : 0;
  const float* hrow = h + (size_t)em * DOUT;
  const float* xrow = x + (size_t)src[em] * DIN;
  const _Float16* bp = w2s + (size_t)(lane & 15) * 32 + bk0;
  v8f zero = {};
  v8f acc[NT];
#pragma unroll
  for (int t = 0; t < NT; ++t) acc[t] = zero;

  for (int k0 = 0; k0 < KP; k0 += 32) {
    // B fragment loads first: latency overlaps the A-generation VALU chain.
    v8h b0[NT], b1[NT];
#pragma unroll
    for (int t = 0; t < NT; ++t) {
      b0[t] = *(const v8h*)(bp + t * 512);
      b1[t] = *(const v8h*)(bp + t * 512 + 16);
    }
    bp += (size_t)DOUT * 32;

    v16h a;
    if constexpr (DIN >= 32) {
      if (k0 < DIN * DOUT) {               // uniform (scalar) branch
        int k = k0 / DIN;
        int ib = k0 - k * DIN;             // 32-aligned within row
        float hv = hrow[k];
#pragma unroll
        for (int j = 0; j < 8; ++j) {
          a[j]     = (_Float16)(hv * xrow[ib + ah + j]);
          a[8 + j] = (_Float16)(hv * xrow[ib + 16 + ah + j]);
        }
      } else {                             // bias rows: virtual h == 1
        int ib = k0 - DIN * DOUT;
#pragma unroll
        for (int j = 0; j < 8; ++j) {
          a[j]     = (_Float16)xrow[ib + ah + j];
          a[8 + j] = (_Float16)xrow[ib + 16 + ah + j];
        }
      }
    } else {
#pragma unroll
      for (int j = 0; j < 16; ++j) {
        int kk = k0 + (j < 8 ? ah + j : 16 + ah + (j - 8));
        int kc = imin(kk / DIN, DOUT - 1);
        int ic = imin(kk - kc * DIN, DIN - 1);
        float av_w = hrow[kc] * xrow[ic];
        int ibc = imin(kk - DIN * DOUT < 0 ? 0 : kk - DIN * DOUT, DIN - 1);
        float av_b = xrow[ibc];
        float av = (kk < DIN * DOUT) ? av_w
                   : (kk < DIN * DOUT + DIN) ? av_b : 0.0f;
        a[j] = (_Float16)av;
      }
    }
#pragma unroll
    for (int t = 0; t < NT; ++t) {
      v16h b = cat8(b0[t], b1[t]);
      acc[t] = __builtin_amdgcn_wmma_f32_16x16x32_f16(false, a, false, b,
                                                      (short)0, acc[t], false,
                                                      false);
    }
  }
#pragma unroll
  for (int t = 0; t < NT; ++t)
#pragma unroll
    for (int r = 0; r < 8; ++r)
      msg[(size_t)(e0 + r + ah) * DOUT + t * 16 + (lane & 15)] = acc[t][r];
}

// x_out[t] = mean_{40 in-edges} msg[e] + (x@root)[t] + bias.
// Target t=g*40+jj owns edges e = g*1600 + ii*40 + jj.  One wave per node.
template <int DOUT>
__global__ void nnconv_combine(const float* __restrict__ msg,
                               const float* __restrict__ rootb,
                               const float* __restrict__ bias,
                               float* __restrict__ xout) {
  constexpr int CPL = DOUT / 32;
  int t = blockIdx.x, lane = threadIdx.x;
  int g = t / NPG, jj = t - g * NPG;
  int ebase = g * NPG * NPG + jj;
  float s[CPL];
#pragma unroll
  for (int c = 0; c < CPL; ++c) s[c] = 0.0f;
  for (int ii = 0; ii < NPG; ++ii) {
    const float* mrow = msg + (size_t)(ebase + ii * NPG) * DOUT;
#pragma unroll
    for (int c = 0; c < CPL; ++c) s[c] += mrow[lane + 32 * c];
  }
#pragma unroll
  for (int c = 0; c < CPL; ++c) {
    int ch = lane + 32 * c;
    xout[(size_t)t * DOUT + ch] =
        s[c] * (1.0f / NPG) + rootb[(size_t)t * DOUT + ch] + bias[ch];
  }
}

// alpha[e] = dot(q[tgt], k[src] + ef[e]) / 16.  One wave per edge.
__global__ void attn_alpha(const float* __restrict__ q, const float* __restrict__ k,
                           const float* __restrict__ ef, const int* __restrict__ src,
                           const int* __restrict__ tgt, float* __restrict__ alpha) {
  int e = blockIdx.x, lane = threadIdx.x;
  const float* qr = q + (size_t)tgt[e] * HDIM;
  const float* kr = k + (size_t)src[e] * HDIM;
  const float* er = ef + (size_t)e * HDIM;
  float acc = 0.0f;
#pragma unroll
  for (int c = 0; c < HDIM / 32; ++c) {
    int ch = lane + 32 * c;
    acc += qr[ch] * (kr[ch] + er[ch]);
  }
  acc = wave_sum32(acc);
  if (lane == 0) alpha[e] = acc * 0.0625f;  // 1/sqrt(256)
}

// Scatter-softmax over each node's 40 in-edges + weighted sum of v[src]+ef.
__global__ void attn_agg(const float* __restrict__ alpha, const float* __restrict__ v,
                         const float* __restrict__ ef, const int* __restrict__ src,
                         float* __restrict__ out) {
  int t = blockIdx.x, lane = threadIdx.x;
  int g = t / NPG, jj = t - g * NPG;
  int ebase = g * NPG * NPG + jj;
  float m = -3.4e38f;
  for (int ii = lane; ii < NPG; ii += 32) m = fmaxf(m, alpha[ebase + ii * NPG]);
  m = wave_max32(m);
  float ssum = 0.0f;
  for (int ii = lane; ii < NPG; ii += 32) ssum += __expf(alpha[ebase + ii * NPG] - m);
  ssum = wave_sum32(ssum);
  float inv = 1.0f / ssum;
  float acc[HDIM / 32];
#pragma unroll
  for (int c = 0; c < HDIM / 32; ++c) acc[c] = 0.0f;
  for (int ii = 0; ii < NPG; ++ii) {
    int e = ebase + ii * NPG;
    float a = __expf(alpha[e] - m) * inv;
    const float* vr = v + (size_t)src[e] * HDIM;
    const float* er = ef + (size_t)e * HDIM;
#pragma unroll
    for (int c = 0; c < HDIM / 32; ++c) {
      int ch = lane + 32 * c;
      acc[c] += a * (vr[ch] + er[ch]);
    }
  }
#pragma unroll
  for (int c = 0; c < HDIM / 32; ++c)
    out[(size_t)t * HDIM + lane + 32 * c] = acc[c];
}

// x = gelu(layernorm(attout + skip(x)+skip.b + x) * g + b).  One wave per node.
__global__ void resid_ln_gelu(const float* __restrict__ attout,
                              const float* __restrict__ skipb,
                              const float* __restrict__ xin,
                              const float* __restrict__ g,
                              const float* __restrict__ b,
                              float* __restrict__ xout) {
  int n = blockIdx.x, lane = threadIdx.x;
  float vals[HDIM / 32];
  float s1 = 0.0f, s2 = 0.0f;
#pragma unroll
  for (int c = 0; c < HDIM / 32; ++c) {
    int ch = lane + 32 * c;
    size_t idx = (size_t)n * HDIM + ch;
    float vv = attout[idx] + skipb[idx] + xin[idx];
    vals[c] = vv;
    s1 += vv;
    s2 += vv * vv;
  }
  s1 = wave_sum32(s1);
  s2 = wave_sum32(s2);
  float mu = s1 * (1.0f / HDIM);
  float var = s2 * (1.0f / HDIM) - mu * mu;
  float r = rsqrtf(var + 1e-5f);
#pragma unroll
  for (int c = 0; c < HDIM / 32; ++c) {
    int ch = lane + 32 * c;
    xout[(size_t)n * HDIM + ch] = gelu_tanh((vals[c] - mu) * r * g[ch] + b[ch]);
  }
}

// Global mean pool per graph + linear head.  One wave per graph.
__global__ void pool_head(const float* __restrict__ x, const float* __restrict__ hw,
                          const float* __restrict__ hb, float* __restrict__ out) {
  int gph = blockIdx.x, lane = threadIdx.x;
  float acc = 0.0f;
#pragma unroll
  for (int c = 0; c < HDIM / 32; ++c) {
    int ch = lane + 32 * c;
    float s = 0.0f;
    for (int j = 0; j < NPG; ++j)
      s += x[(size_t)(gph * NPG + j) * HDIM + ch];
    acc += s * (1.0f / NPG) * hw[ch];
  }
  acc = wave_sum32(acc);
  if (lane == 0) out[gph] = acc + hb[0];
}

// ---------------------------------------------------------------------------
// Host orchestration
// ---------------------------------------------------------------------------
extern "C" void kernel_launch(void* const* d_in, const int* in_sizes, int n_in,
                              void* d_out, int out_size, void* d_ws, size_t ws_size,
                              hipStream_t stream) {
  (void)n_in; (void)out_size; (void)ws_size;
  const float* x_in = (const float*)d_in[0];
  const int*   eidx = (const int*)d_in[1];
  const int*   src  = eidx;
  const int*   tgt  = eidx + EE;
  const float* ea   = (const float*)d_in[2];

  // Parameter leaf ordering: sorted-key vs insertion-order pytree flattening,
  // discriminated by in_sizes[4] (gnn0.bias = 32 vs gnn0.mlp1.w = 256).
  bool sorted_order = (in_sizes[4] == 32);

  const float *gnn_bias[3], *gnn_m1w[3], *gnn_m1b[3], *gnn_m2w[3], *gnn_m2b[3],
      *gnn_rw[3];
  const float *projw, *projb, *headw, *headb;
  const float *t_ew[6], *t_kw[6], *t_kb[6], *t_qw[6], *t_qb[6], *t_vw[6],
      *t_vb[6], *t_sw[6], *t_sb[6], *t_lng[6], *t_lnb[6];
  auto f = [&](int i) { return (const float*)d_in[i]; };

  if (sorted_order) {
    for (int L = 0; L < 3; ++L) {
      int base = 4 + 6 * L;
      gnn_bias[L] = f(base + 0); gnn_m1b[L] = f(base + 1); gnn_m1w[L] = f(base + 2);
      gnn_m2b[L] = f(base + 3); gnn_m2w[L] = f(base + 4); gnn_rw[L] = f(base + 5);
    }
    headb = f(22); headw = f(23); projb = f(24); projw = f(25);
    for (int t = 0; t < 6; ++t) {
      int base = 26 + 11 * t;
      t_ew[t] = f(base + 0); t_kb[t] = f(base + 1); t_kw[t] = f(base + 2);
      t_lnb[t] = f(base + 3); t_lng[t] = f(base + 4); t_qb[t] = f(base + 5);
      t_qw[t] = f(base + 6); t_sb[t] = f(base + 7); t_sw[t] = f(base + 8);
      t_vb[t] = f(base + 9); t_vw[t] = f(base + 10);
    }
  } else {
    for (int L = 0; L < 3; ++L) {
      int base = 4 + 6 * L;
      gnn_m1w[L] = f(base + 0); gnn_m1b[L] = f(base + 1); gnn_m2w[L] = f(base + 2);
      gnn_m2b[L] = f(base + 3); gnn_rw[L] = f(base + 4); gnn_bias[L] = f(base + 5);
    }
    projw = f(22); projb = f(23);
    for (int t = 0; t < 6; ++t) {
      int base = 24 + 11 * t;
      t_qw[t] = f(base + 0); t_qb[t] = f(base + 1); t_kw[t] = f(base + 2);
      t_kb[t] = f(base + 3); t_vw[t] = f(base + 4); t_vb[t] = f(base + 5);
      t_ew[t] = f(base + 6); t_sw[t] = f(base + 7); t_sb[t] = f(base + 8);
      t_lng[t] = f(base + 9); t_lnb[t] = f(base + 10);
    }
    headw = f(90); headb = f(91);
  }

  // ---- workspace carving (offsets 256B-aligned)
  char* ws = (char*)d_ws;
  float*    xa      = (float*)(ws + 0);           // 640*256*4
  float*    xb      = (float*)(ws + 655360);
  float*    qbuf    = (float*)(ws + 1310720);     // doubles as rootbuf in GNN phase
  float*    kbuf    = (float*)(ws + 1966080);
  float*    vbuf    = (float*)(ws + 2621440);
  float*    skipbuf = (float*)(ws + 3276800);
  float*    attbuf  = (float*)(ws + 3932160);
  float*    alphab  = (float*)(ws + 4587520);     // 25600*4
  float*    hbuf    = (float*)(ws + 4689920);     // 25600*128*4
  float*    msgbuf  = (float*)(ws + 17797120);    // 25600*128*4
  float*    efbuf   = (float*)(ws + 30904320);    // 25600*256*4
  _Float16* w2s     = (_Float16*)(ws + 57118720); // 8256*128*2
  _Float16* ea16    = (_Float16*)(ws + 59232256); // 25600*32*2
  _Float16* act16   = (_Float16*)(ws + 60870656); // 640*256*2 (A staging)
  _Float16* wb16    = (_Float16*)(ws + 61198336); // 256*256*2 (B staging)

  dim3 wave(32);
  auto packA = [&](const float* A, _Float16* Ap, int M, int K, int KP) {
    int tot = M * KP;
    pack_a<<<(tot + 255) / 256, 256, 0, stream>>>(A, Ap, M, K, KP);
  };
  auto packB = [&](const float* B, _Float16* Bp, int K, int N, int KP) {
    int tot = KP * N;
    pack_b<<<(tot + 255) / 256, 256, 0, stream>>>(B, Bp, K, N, KP);
  };
  auto gemm = [&](const _Float16* A16, const _Float16* B16, const float* bias,
                  float* C, int M, int N, int KP, int act) {
    if (N % 64 == 0) {
      dim3 grid(N / 64, M / 16);
      wmma_gemm_f16<4><<<grid, wave, 0, stream>>>(A16, B16, bias, C, M, N, KP, act);
    } else {
      dim3 grid(N / 32, M / 16);
      wmma_gemm_f16<2><<<grid, wave, 0, stream>>>(A16, B16, bias, C, M, N, KP, act);
    }
  };

  // edge_attr staged once: [25600, 8] -> f16 [25600, 32] (reused 9x)
  packA(ea, ea16, EE, EDGE_DIM, 32);

#define GNN_LAYER(DIN, DINP, DOUT, KP, XIN, XOUT, L)                               \
  do {                                                                             \
    packB(gnn_m1w[L], wb16, EDGE_DIM, DOUT, 32);                                   \
    gemm(ea16, wb16, gnn_m1b[L], hbuf, EE, DOUT, 32, 1); /* h=gelu(EA@W1+b1) */    \
    prep_w2<DIN, DOUT, KP>                                                         \
        <<<((KP) * (DOUT) + 255) / 256, 256, 0, stream>>>(gnn_m2w[L],              \
                                                          gnn_m2b[L], w2s);        \
    nnconv_msg<DIN, DOUT, KP><<<EE / 16, wave, 0, stream>>>(hbuf, XIN, src, w2s,   \
                                                            msgbuf);               \
    packA(XIN, act16, NN, DIN, DINP);                                              \
    packB(gnn_rw[L], wb16, DIN, DOUT, DINP);                                       \
    gemm(act16, wb16, nullptr, qbuf, NN, DOUT, DINP, 0);                           \
    nnconv_combine<DOUT><<<NN, wave, 0, stream>>>(msgbuf, qbuf, gnn_bias[L],       \
                                                  XOUT);                           \
  } while (0)

  GNN_LAYER(7, 32, 32, 256, x_in, xa, 0);
  GNN_LAYER(32, 32, 64, 2080, xa, xb, 1);
  GNN_LAYER(64, 64, 128, 8256, xb, xa, 2);
#undef GNN_LAYER

  // projection 128 -> 256
  packA(xa, act16, NN, 128, 128);
  packB(projw, wb16, 128, HDIM, 128);
  gemm(act16, wb16, projb, xb, NN, HDIM, 128, 0);

  float* cur = xb;
  float* nxt = xa;
  for (int t = 0; t < 6; ++t) {
    packA(cur, act16, NN, HDIM, HDIM);
    packB(t_qw[t], wb16, HDIM, HDIM, HDIM);
    gemm(act16, wb16, t_qb[t], qbuf, NN, HDIM, HDIM, 0);
    packB(t_kw[t], wb16, HDIM, HDIM, HDIM);
    gemm(act16, wb16, t_kb[t], kbuf, NN, HDIM, HDIM, 0);
    packB(t_vw[t], wb16, HDIM, HDIM, HDIM);
    gemm(act16, wb16, t_vb[t], vbuf, NN, HDIM, HDIM, 0);
    packB(t_sw[t], wb16, HDIM, HDIM, HDIM);
    gemm(act16, wb16, t_sb[t], skipbuf, NN, HDIM, HDIM, 0);
    packB(t_ew[t], wb16, EDGE_DIM, HDIM, 32);
    gemm(ea16, wb16, nullptr, efbuf, EE, HDIM, 32, 0);
    attn_alpha<<<EE, wave, 0, stream>>>(qbuf, kbuf, efbuf, src, tgt, alphab);
    attn_agg<<<NN, wave, 0, stream>>>(alphab, vbuf, efbuf, src, attbuf);
    resid_ln_gelu<<<NN, wave, 0, stream>>>(attbuf, skipbuf, cur, t_lng[t],
                                           t_lnb[t], nxt);
    float* tmp = cur; cur = nxt; nxt = tmp;
  }

  pool_head<<<N_GRAPHS, wave, 0, stream>>>(cur, headw, headb, (float*)d_out);
}
